// BigBirdProteinModel_37907381354603
// MI455X (gfx1250) — matrix operand order repaired and checked by reading, not compiled, not run on hardware
//
#include <hip/hip_runtime.h>

typedef __attribute__((ext_vector_type(8)))  __bf16 bf16x8;
typedef __attribute__((ext_vector_type(16))) __bf16 v16bf;
typedef __attribute__((ext_vector_type(8)))  float  v8f;

namespace cfg {
constexpr int B = 2, S = 4096, IN = 1280, D = 768, H = 12, HD = 64;
constexpr int L = 12, FF = 3072, TGT = 1024, NR = 3, NB = 64;
constexpr int M = B * S; // 8192
}

#define GF_F32 1
#define GF_BF16 2
#define GF_GELU 4
#define GF_RES 8
#define GF_VT 16

#if __has_builtin(__builtin_amdgcn_global_load_async_to_lds_b128)
#define HAVE_ASYNC_BUILTIN 1
#else
#define HAVE_ASYNC_BUILTIN 0
#endif

#if __has_builtin(__builtin_amdgcn_tensor_load_to_lds)
#define HAVE_TDM 1
#else
#define HAVE_TDM 0
#endif

// ------------- async global->LDS copy (ASYNCcnt path) -------------
// Builtin signature (from clang diagnostic): param0 is a pointer to
// vector_size(16) int in addrspace(1); param1 is the LDS-side pointer.
// LDS byte address = low 32 bits of the flat address of a __shared__ object
// (aperture lives in bits 63:32), so integer casts give us the AS(3) value.

typedef int v4i_gcc __attribute__((vector_size(16)));
typedef __attribute__((address_space(1))) v4i_gcc* as1_v4i_ptr;
typedef __attribute__((address_space(3))) v4i_gcc* as3_v4i_ptr;

__device__ __forceinline__ void async_b128(void* l, const void* g) {
#if HAVE_ASYNC_BUILTIN
  __builtin_amdgcn_global_load_async_to_lds_b128(
      (as1_v4i_ptr)(unsigned long long)g,
      (as3_v4i_ptr)(unsigned)(unsigned long long)l, 0, 0);
#else
  asm volatile("global_load_async_to_lds_b128 %0, %1, off" ::"v"(
                   (unsigned)(unsigned long long)l),
               "v"(g)
               : "memory");
#endif
}

template <int N>
__device__ __forceinline__ void wait_async() {
#if __has_builtin(__builtin_amdgcn_s_wait_asynccnt)
  __builtin_amdgcn_s_wait_asynccnt((short)N);
#else
  asm volatile("s_wait_asynccnt %0" ::"i"(N) : "memory");
#endif
}

template <int N>
__device__ __forceinline__ void wait_tensor() {
#if __has_builtin(__builtin_amdgcn_s_wait_tensorcnt)
  __builtin_amdgcn_s_wait_tensorcnt((short)N);
#else
  asm volatile("s_wait_tensorcnt %0" ::"i"(N) : "memory");
#endif
}

// ------------- TDM tile load (TENSORcnt path) -------------
// 2D tile of bf16: tile_k_el columns x tile_rows rows, global row stride
// row_stride_el, packed into LDS with 16B pad per 64B row (pitch = 40 elems).

#if HAVE_TDM
typedef unsigned int u32x4 __attribute__((ext_vector_type(4)));
typedef int i32x4 __attribute__((ext_vector_type(4)));
typedef int i32x8 __attribute__((ext_vector_type(8)));

__device__ __forceinline__ void tdm_load_tile_2d(unsigned lds_byte_addr,
                                                 const void* gaddr, int tile_k_el,
                                                 int tile_rows,
                                                 long row_stride_el) {
  unsigned long long ga = (unsigned long long)gaddr;
  u32x4 g0;
  g0[0] = 1u;                 // count=1, user descriptor
  g0[1] = lds_byte_addr;      // lds_addr
  g0[2] = (unsigned)ga;       // global_addr[31:0]
  g0[3] = (unsigned)((ga >> 32) & 0x01ffffffu) | (2u << 30);  // type=2 (image)
  unsigned td0 = 1u << 20, td1 = 1u << 20;  // huge tensor dims: no OOB clamp
  unsigned long long st = (unsigned long long)row_stride_el;
  i32x8 g1;
  // data_size=2B (bits17:16=1), pad_enable (bit20), pad_interval=16DW (3<<22),
  // pad_amount=4DW (3<<25); workgroup_mask=0 (not in a cluster)
  g1[0] = (int)((1u << 16) | (1u << 20) | (3u << 22) | (3u << 25));
  g1[1] = (int)((td0 & 0xffffu) << 16);                     // tensor_dim0 lo
  g1[2] = (int)((td0 >> 16) | ((td1 & 0xffffu) << 16));     // dim0 hi | dim1 lo
  g1[3] = (int)((td1 >> 16) | ((unsigned)tile_k_el << 16)); // dim1 hi | tile0
  g1[4] = (int)(unsigned)tile_rows;                         // tile1 (tile2=0)
  g1[5] = (int)(unsigned)(st & 0xffffffffull);              // dim0_stride lo
  g1[6] = (int)(unsigned)(st >> 32);                        // dim0_stride hi
  g1[7] = 0;
  i32x4 gz = {0, 0, 0, 0};
#if __clang_major__ >= 23
  i32x8 gz8 = {0, 0, 0, 0, 0, 0, 0, 0};
  __builtin_amdgcn_tensor_load_to_lds(g0, g1, gz, gz, gz8, 0);
#else
  __builtin_amdgcn_tensor_load_to_lds(g0, g1, gz, gz, 0);
#endif
}
constexpr int GEMM_STAGE_ASYNC = 1;  // only B tile uses ASYNCcnt
#else
constexpr int GEMM_STAGE_ASYNC = 3;  // A (2) + B (1)
#endif

// ---------------- WMMA helpers (CDNA5 wave32, 16x16x32 bf16) ----------------

__device__ __forceinline__ v8f wmma_bf16(v16bf a, v16bf b, v8f c) {
  return __builtin_amdgcn_wmma_f32_16x16x32_bf16(false, a, false, b, (short)0, c,
                                                 false, false);
}

// A 16x32 fragment: lanes 0-15 row M=lane (K=half*8+0..7, 16+half*8+0..7)
__device__ __forceinline__ v16bf frag_a(const __bf16* s, int row0, int k0, int st,
                                        int lane) {
  int m = lane & 15, hf = lane >> 4;
  const __bf16* p = s + (row0 + m) * st + k0 + hf * 8;
  bf16x8 lo = *(const bf16x8*)p;
  bf16x8 hi = *(const bf16x8*)(p + 16);
  return __builtin_shufflevector(lo, hi, 0, 1, 2, 3, 4, 5, 6, 7, 8, 9, 10, 11, 12,
                                 13, 14, 15);
}

// B 32x16 fragment from transposed tile t[n][k]: lane col N=lane&15,
// K = half*8+0..7, 16+half*8+0..7 (same order as frag_a).
__device__ __forceinline__ v16bf frag_b(const __bf16* t, int k0, int n0, int st,
                                        int lane) {
  int n = lane & 15, hf = lane >> 4;
  const __bf16* p = t + (n0 + n) * st + k0 + hf * 8;
  bf16x8 lo = *(const bf16x8*)p;
  bf16x8 hi = *(const bf16x8*)(p + 16);
  return __builtin_shufflevector(lo, hi, 0, 1, 2, 3, 4, 5, 6, 7, 8, 9, 10, 11, 12,
                                 13, 14, 15);
}

// ---------------- fp32 -> bf16 conversions ----------------

__global__ void k_cvt(const float* __restrict__ s, __bf16* __restrict__ d, int n) {
  int i = blockIdx.x * 256 + threadIdx.x;
  if (i < n) d[i] = (__bf16)s[i];
}

// transpose-convert: src[K,N] fp32 -> dst[N,K] bf16
__global__ void k_cvtT(const float* __restrict__ s, __bf16* __restrict__ d, int K,
                       int N) {
  int i = blockIdx.x * 256 + threadIdx.x;
  if (i >= K * N) return;
  int n = i / K, k = i - n * K;
  d[i] = (__bf16)s[(long)k * N + n];
}

// ---------------- Tiled GEMM: out = A[M,K] @ WT[N,K]^T + bias ----------------
// 256 threads = 8 waves, tile 128x64, K-step 32, double-buffered async staging.

__global__ __launch_bounds__(256) void k_gemm(
    const __bf16* __restrict__ A, const __bf16* __restrict__ WT,
    const float* __restrict__ bias, const float* __restrict__ res,
    float* __restrict__ outF, __bf16* __restrict__ outB, int Mm, int Nn, int Kk,
    int flags) {
  (void)Mm;
  __shared__ __bf16 As[2][128 * 40];
  __shared__ __bf16 Bs[2][64 * 40];
  int t = threadIdx.x, lane = t & 31, w = t >> 5;
  int n0 = blockIdx.x * 64;
  long m0 = (long)blockIdx.y * 128;
  v8f acc[4] = {};

  auto stage = [&](int buf, int k0) {
    // B tile: 64x32 contiguous rows of WT -> async b128, 1 instr/thread
    int br = t >> 2, bc = (t & 3) * 8;
    async_b128(&Bs[buf][br * 40 + bc], WT + (long)(n0 + br) * Kk + k0 + bc);
#if HAVE_TDM
    // A tile: one TDM descriptor (wave 0 issues; EXEC-independent DMA)
    if (t < 32)
      tdm_load_tile_2d((unsigned)(unsigned long long)&As[buf][0],
                       A + m0 * Kk + k0, 32, 128, Kk);
#else
#pragma unroll
    for (int j = 0; j < 2; ++j) {
      int grp = t + j * 256;
      int row = grp >> 2, cg = (grp & 3) * 8;
      async_b128(&As[buf][row * 40 + cg], A + (m0 + row) * Kk + k0 + cg);
    }
#endif
  };

  stage(0, 0);
  int nsteps = Kk / 32;
  for (int st = 0; st < nsteps; ++st) {
    int buf = st & 1;
    bool more = (st + 1) < nsteps;
    if (more) stage(buf ^ 1, (st + 1) * 32);
    if (more) wait_async<GEMM_STAGE_ASYNC>(); else wait_async<0>();
#if HAVE_TDM
    if (more) wait_tensor<1>(); else wait_tensor<0>();
#endif
    __syncthreads();
    v16bf a = frag_a(&As[buf][0], w * 16, 0, 40, lane);
#pragma unroll
    for (int nt = 0; nt < 4; ++nt)
      acc[nt] = wmma_bf16(a, frag_b(&Bs[buf][0], 0, nt * 16, 40, lane), acc[nt]);
    __syncthreads();
  }
  // epilogue: C row = i + 8*(lane>=16), col = lane&15
  int hf = lane >> 4, cl = lane & 15;
#pragma unroll
  for (int nt = 0; nt < 4; ++nt) {
    int col = n0 + nt * 16 + cl;
    float bv = bias ? bias[col] : 0.f;
#pragma unroll
    for (int i = 0; i < 8; ++i) {
      long row = m0 + w * 16 + i + 8 * hf;
      float x = acc[nt][i] + bv;
      if (flags & GF_RES) x += res[row * Nn + col];
      if (flags & GF_GELU) {
        float u = x;
        x = 0.5f * u *
            (1.f + tanhf(0.7978845608028654f * (u + 0.044715f * u * u * u)));
      }
      if (flags & GF_F32) outF[row * Nn + col] = x;
      if (flags & GF_VT) {
        // scatter V as VT[(b*H+h)*HD+hd][S] for async attention staging
        int bI = (int)(row >> 12), sI = (int)(row & 4095);
        int hI = col >> 6, hd = col & 63;
        long vt = (((long)bI * cfg::H + hI) * cfg::HD + hd) * cfg::S + sI;
        outB[vt] = (__bf16)x;
      } else if (flags & GF_BF16) {
        outB[row * Nn + col] = (__bf16)x;
      }
    }
  }
}

// ---------------- LayerNorm (optionally fused +pos_emb+tok_emb) --------------

__global__ __launch_bounds__(256) void k_ln(const float* __restrict__ in,
                                            const float* __restrict__ pos,
                                            const float* __restrict__ tok,
                                            const float* __restrict__ gma,
                                            const float* __restrict__ bta,
                                            float* __restrict__ outF,
                                            __bf16* __restrict__ outB) {
  constexpr int D = cfg::D;
  long row = blockIdx.x;
  int t = threadIdx.x;
  int srow = (int)(row % cfg::S);
  __shared__ float r1[256], r2[256];
  float vals[4];
  float sum = 0.f, sq = 0.f;
  int c = 0;
  for (int i = t; i < D; i += 256) {
    float x = in[row * D + i];
    if (pos) x += pos[(long)srow * D + i];
    if (tok) x += tok[i];
    vals[c++] = x;
    sum += x;
    sq += x * x;
  }
  r1[t] = sum;
  r2[t] = sq;
  __syncthreads();
  for (int s2 = 128; s2 > 0; s2 >>= 1) {
    if (t < s2) {
      r1[t] += r1[t + s2];
      r2[t] += r2[t + s2];
    }
    __syncthreads();
  }
  float mean = r1[0] / D;
  float var = r2[0] / D - mean * mean;
  float inv = rsqrtf(var + 1e-12f);
  c = 0;
  for (int i = t; i < D; i += 256) {
    float y = (vals[c++] - mean) * inv * gma[i] + bta[i];
    if (outF) outF[row * D + i] = y;
    if (outB) outB[row * D + i] = (__bf16)y;
  }
}

// ---------- BigBird block-sparse attention (flash-style, WMMA, async) --------
// grid (NB, H, B), 128 threads = 4 waves; double-buffered K/V staging.

__global__ __launch_bounds__(128) void k_attn(const __bf16* __restrict__ Qg,
                                              const __bf16* __restrict__ Kg,
                                              const __bf16* __restrict__ VT,
                                              const float* __restrict__ msk,
                                              const int* __restrict__ rb,
                                              __bf16* __restrict__ ctx) {
  using namespace cfg;
  __shared__ __bf16 qs[64 * 72];       // [q][hd]
  __shared__ __bf16 ks[2][64 * 72];    // [key][hd] (transposed-B for QK^T)
  __shared__ __bf16 vts[2][64 * 72];   // [hd][key] (transposed-B for P@V)
  __shared__ __bf16 ps[64 * 72];       // [q][key] P tile
  int t = threadIdx.x, lane = t & 31, w = t >> 5;
  int qb = blockIdx.x, hh = blockIdx.y, b = blockIdx.z;
  long rowbase = (long)b * S + (long)qb * 64;
  int cbase = hh * HD;
  const __bf16* vthead = VT + ((long)(b * H + hh) * HD) * S;

  // Q tile: 4 async instructions
#pragma unroll
  for (int j = 0; j < 4; ++j) {
    int grp = t + j * 128;
    int row = grp >> 3, c0 = (grp & 7) * 8;
    async_b128(&qs[row * 72 + c0], Qg + (rowbase + row) * D + cbase + c0);
  }

  auto stage_kv = [&](int buf, int kb) {  // 8 async instructions
    long kbase = (long)b * S + (long)kb * 64;
#pragma unroll
    for (int jj = 0; jj < 4; ++jj) {
      int grp = t + jj * 128;
      int row = grp >> 3, c0 = (grp & 7) * 8;
      async_b128(&ks[buf][row * 72 + c0], Kg + (kbase + row) * D + cbase + c0);
      async_b128(&vts[buf][row * 72 + c0],
                 vthead + (long)row * S + (long)kb * 64 + c0);
    }
  };

  bool edge = (qb == 0) || (qb == NB - 1);
  int list[8];
  int nkb = edge ? NB : 8;
  if (!edge) {
    list[0] = 0;
    list[1] = qb - 1;
    list[2] = qb;
    list[3] = qb + 1;
    list[4] = NB - 1;
    list[5] = rb[qb * NR + 0];
    list[6] = rb[qb * NR + 1];
    list[7] = rb[qb * NR + 2];
  }
  float mstat[8], lstat[8];
#pragma unroll
  for (int i = 0; i < 8; ++i) {
    mstat[i] = -3.0e38f;
    lstat[i] = 0.f;
  }
  v8f of[4] = {};
  stage_kv(0, edge ? 0 : list[0]);
  for (int j = 0; j < nkb; ++j) {
    int buf = j & 1;
    bool more = (j + 1) < nkb;
    if (more) stage_kv(buf ^ 1, edge ? (j + 1) : list[j + 1]);
    if (more) wait_async<8>(); else wait_async<0>();
    __syncthreads();
    int kb = edge ? j : list[j];
    // S = Q @ Kb^T  (HD=64 -> two wmma K-steps)
    v8f sf[4] = {};
    v16bf a0 = frag_a(qs, w * 16, 0, 72, lane);
    v16bf a1 = frag_a(qs, w * 16, 32, 72, lane);
#pragma unroll
    for (int nt = 0; nt < 4; ++nt) {
      sf[nt] = wmma_bf16(a0, frag_b(&ks[buf][0], 0, nt * 16, 72, lane), sf[nt]);
      sf[nt] = wmma_bf16(a1, frag_b(&ks[buf][0], 32, nt * 16, 72, lane), sf[nt]);
    }
    // scale + mask; per-lane rows are i + 8*(lane>=16)
    float rm[8];
#pragma unroll
    for (int i = 0; i < 8; ++i) rm[i] = -3.0e38f;
#pragma unroll
    for (int nt = 0; nt < 4; ++nt) {
      int col = nt * 16 + (lane & 15);
      float mv = msk[(long)b * S + (long)kb * 64 + col];
      float ng = -1.0e9f * (1.f - mv);
#pragma unroll
      for (int i = 0; i < 8; ++i) {
        float sv = sf[nt][i] * 0.125f + ng;
        sf[nt][i] = sv;
        rm[i] = fmaxf(rm[i], sv);
      }
    }
#pragma unroll
    for (int i = 0; i < 8; ++i) {
#pragma unroll
      for (int o = 1; o < 16; o <<= 1)
        rm[i] = fmaxf(rm[i], __shfl_xor(rm[i], o, 32));
    }
    float fac[8], rs[8];
#pragma unroll
    for (int i = 0; i < 8; ++i) {
      float mn = fmaxf(mstat[i], rm[i]);
      fac[i] = expf(mstat[i] - mn);
      mstat[i] = mn;
      rs[i] = 0.f;
    }
#pragma unroll
    for (int nt = 0; nt < 4; ++nt) {
#pragma unroll
      for (int i = 0; i < 8; ++i) {
        float p = expf(sf[nt][i] - mstat[i]);
        sf[nt][i] = p;
        rs[i] += p;
      }
    }
#pragma unroll
    for (int i = 0; i < 8; ++i) {
#pragma unroll
      for (int o = 1; o < 16; o <<= 1) rs[i] += __shfl_xor(rs[i], o, 32);
      lstat[i] = lstat[i] * fac[i] + rs[i];
    }
    {
      int hf = lane >> 4, cl = lane & 15;
#pragma unroll
      for (int nt = 0; nt < 4; ++nt) {
        int col = nt * 16 + cl;
#pragma unroll
        for (int i = 0; i < 8; ++i)
          ps[(w * 16 + i + 8 * hf) * 72 + col] = (__bf16)sf[nt][i];
      }
    }
#pragma unroll
    for (int nt = 0; nt < 4; ++nt) {
#pragma unroll
      for (int i = 0; i < 8; ++i) of[nt][i] *= fac[i];
    }
    // O += P @ Vb
    v16bf p0 = frag_a(ps, w * 16, 0, 72, lane);
    v16bf p1 = frag_a(ps, w * 16, 32, 72, lane);
#pragma unroll
    for (int nt = 0; nt < 4; ++nt) {
      of[nt] = wmma_bf16(p0, frag_b(&vts[buf][0], 0, nt * 16, 72, lane), of[nt]);
      of[nt] = wmma_bf16(p1, frag_b(&vts[buf][0], 32, nt * 16, 72, lane), of[nt]);
    }
    __syncthreads();
  }
  int hf = lane >> 4, cl = lane & 15;
#pragma unroll
  for (int nt = 0; nt < 4; ++nt) {
    int col = cbase + nt * 16 + cl;
#pragma unroll
    for (int i = 0; i < 8; ++i) {
      long row = rowbase + w * 16 + i + 8 * hf;
      float o = of[nt][i] / lstat[i];
      ctx[row * D + col] = (__bf16)o;
    }
  }
}

// ---------------- tiny classifier head (fp32) ----------------

__global__ void k_cls1(const float* __restrict__ h, const float* __restrict__ w1,
                       const float* __restrict__ b1, float* __restrict__ t1) {
  int n = blockIdx.x * 256 + threadIdx.x;
  int bb = blockIdx.y;
  if (n >= 512) return;
  const float* row = h + (long)bb * cfg::S * cfg::D;  // token 0
  float acc = b1[n];
  for (int k = 0; k < cfg::D; ++k) acc += row[k] * w1[k * 512 + n];
  t1[bb * 512 + n] = fmaxf(acc, 0.f);
}

__global__ void k_cls2(const float* __restrict__ t1, const float* __restrict__ w2,
                       const float* __restrict__ b2, float* __restrict__ out) {
  int n = blockIdx.x * 256 + threadIdx.x;
  int bb = blockIdx.y;
  if (n >= cfg::TGT) return;
  float acc = b2[n];
  for (int k = 0; k < 512; ++k) acc += t1[bb * 512 + k] * w2[k * cfg::TGT + n];
  out[bb * cfg::TGT + n] = acc;
}

// ---------------- host orchestration ----------------

extern "C" void kernel_launch(void* const* d_in, const int* in_sizes, int n_in,
                              void* d_out, int out_size, void* d_ws, size_t ws_size,
                              hipStream_t stream) {
  using namespace cfg;
  (void)in_sizes; (void)n_in; (void)out_size; (void)ws_size;
  const float* x = (const float*)d_in[0];
  const float* amask = (const float*)d_in[1];
  const float* proj_w = (const float*)d_in[2];
  const float* proj_b = (const float*)d_in[3];
  const float* pos_emb = (const float*)d_in[4];
  const float* tok_emb = (const float*)d_in[5];
  const float* embg = (const float*)d_in[6];
  const float* embb = (const float*)d_in[7];
  const float* wq = (const float*)d_in[8];
  const float* bq = (const float*)d_in[9];
  const float* wk = (const float*)d_in[10];
  const float* bk = (const float*)d_in[11];
  const float* wv = (const float*)d_in[12];
  const float* bv = (const float*)d_in[13];
  const float* wo = (const float*)d_in[14];
  const float* bo = (const float*)d_in[15];
  const float* ln1g = (const float*)d_in[16];
  const float* ln1b = (const float*)d_in[17];
  const float* wi = (const float*)d_in[18];
  const float* bi = (const float*)d_in[19];
  const float* wo2 = (const float*)d_in[20];
  const float* bo2 = (const float*)d_in[21];
  const float* ln2g = (const float*)d_in[22];
  const float* ln2b = (const float*)d_in[23];
  const float* cw1 = (const float*)d_in[24];
  const float* cb1 = (const float*)d_in[25];
  const float* cw2 = (const float*)d_in[26];
  const float* cb2 = (const float*)d_in[27];
  const int* rndb = (const int*)d_in[28];

  char* wsp = (char*)d_ws;
  size_t off = 0;
  auto take = [&](size_t bytes) -> void* {
    void* p = wsp + off;
    off = (off + bytes + 255) & ~(size_t)255;
    return p;
  };
  __bf16* xbf = (__bf16*)take((size_t)M * IN * 2);
  __bf16* pwbf = (__bf16*)take((size_t)IN * D * 2);   // [D][IN] transposed
  float* preln = (float*)take((size_t)M * D * 4);
  float* hf32 = (float*)take((size_t)M * D * 4);
  __bf16* hbf = (__bf16*)take((size_t)M * D * 2);
  __bf16* qbf = (__bf16*)take((size_t)M * D * 2);
  __bf16* kbf = (__bf16*)take((size_t)M * D * 2);
  __bf16* vtbf = (__bf16*)take((size_t)M * D * 2);    // V pre-transposed
  __bf16* ctxbf = (__bf16*)take((size_t)M * D * 2);
  __bf16* ffbf = (__bf16*)take((size_t)M * FF * 2);
  __bf16* wql = (__bf16*)take((size_t)D * D * 2);     // all weights [N][K]
  __bf16* wkl = (__bf16*)take((size_t)D * D * 2);
  __bf16* wvl = (__bf16*)take((size_t)D * D * 2);
  __bf16* wol = (__bf16*)take((size_t)D * D * 2);
  __bf16* wil = (__bf16*)take((size_t)D * FF * 2);
  __bf16* wo2l = (__bf16*)take((size_t)FF * D * 2);
  float* t1 = (float*)take((size_t)B * 512 * 4);

  auto cvt = [&](const float* src, __bf16* dst, int n) {
    k_cvt<<<(n + 255) / 256, 256, 0, stream>>>(src, dst, n);
  };
  auto cvtT = [&](const float* src, __bf16* dst, int K, int N) {
    int n = K * N;
    k_cvtT<<<(n + 255) / 256, 256, 0, stream>>>(src, dst, K, N);
  };
  cvt(x, xbf, M * IN);
  cvtT(proj_w, pwbf, IN, D);
  dim3 gD(D / 64, M / 128), gFF(FF / 64, M / 128);
  // h = LN(x @ proj_w + proj_b + pos + tok)
  k_gemm<<<gD, 256, 0, stream>>>(xbf, pwbf, proj_b, nullptr, preln, nullptr, M, D,
                                 IN, GF_F32);
  k_ln<<<M, 256, 0, stream>>>(preln, pos_emb, tok_emb, embg, embb, hf32, hbf);
  for (int l = 0; l < L; ++l) {
    cvtT(wq + (size_t)l * D * D, wql, D, D);
    cvtT(wk + (size_t)l * D * D, wkl, D, D);
    cvtT(wv + (size_t)l * D * D, wvl, D, D);
    cvtT(wo + (size_t)l * D * D, wol, D, D);
    cvtT(wi + (size_t)l * D * FF, wil, D, FF);
    cvtT(wo2 + (size_t)l * FF * D, wo2l, FF, D);
    k_gemm<<<gD, 256, 0, stream>>>(hbf, wql, bq + l * D, nullptr, nullptr, qbf, M,
                                   D, D, GF_BF16);
    k_gemm<<<gD, 256, 0, stream>>>(hbf, wkl, bk + l * D, nullptr, nullptr, kbf, M,
                                   D, D, GF_BF16);
    k_gemm<<<gD, 256, 0, stream>>>(hbf, wvl, bv + l * D, nullptr, nullptr, vtbf, M,
                                   D, D, GF_VT);
    k_attn<<<dim3(NB, H, B), 128, 0, stream>>>(qbf, kbf, vtbf, amask, rndb, ctxbf);
    k_gemm<<<gD, 256, 0, stream>>>(ctxbf, wol, bo + l * D, hf32, preln, nullptr, M,
                                   D, D, GF_F32 | GF_RES);
    k_ln<<<M, 256, 0, stream>>>(preln, nullptr, nullptr, ln1g + l * D,
                                ln1b + l * D, hf32, hbf);
    k_gemm<<<gFF, 256, 0, stream>>>(hbf, wil, bi + l * FF, nullptr, nullptr, ffbf,
                                    M, FF, D, GF_BF16 | GF_GELU);
    k_gemm<<<gD, 256, 0, stream>>>(ffbf, wo2l, bo2 + l * D, hf32, preln, nullptr,
                                   M, D, FF, GF_F32 | GF_RES);
    k_ln<<<M, 256, 0, stream>>>(preln, nullptr, nullptr, ln2g + l * D,
                                ln2b + l * D, hf32, hbf);
  }
  k_cls1<<<dim3(2, B), 256, 0, stream>>>(hf32, cw1, cb1, t1);
  k_cls2<<<dim3(4, B), 256, 0, stream>>>(t1, cw2, cb2, (float*)d_out);
}
